// QuantInternMLP_35124242547299
// MI455X (gfx1250) — 8-bit matrix operand order repaired and checked by reading, compiled, not run on hardware
//
#include <hip/hip_runtime.h>
#include <math.h>


typedef __attribute__((ext_vector_type(4))) int v4i;
typedef __attribute__((ext_vector_type(8))) int v8i;
typedef unsigned long long u64;
typedef unsigned char u8;
typedef unsigned int u32;

#define T_TOK 8192
#define H_DIM 3200
#define I_DIM 12800

#define BLK_M 64
#define BLK_N 128
#define BLK_K 128
#define NTHREADS 256

__device__ __forceinline__ void async_load_b128(u32 lds_addr, const void* gaddr) {
    asm volatile("global_load_async_to_lds_b128 %0, %1, off"
                 :: "v"(lds_addr), "v"(gaddr) : "memory");
}
__device__ __forceinline__ void wait_async_le6() {
    asm volatile("s_wait_asynccnt 0x6" ::: "memory");
}
__device__ __forceinline__ void wait_async_0() {
    asm volatile("s_wait_asynccnt 0x0" ::: "memory");
}

__global__ __launch_bounds__(256) void pack_i8_kernel(const int* __restrict__ in,
                                                      int* __restrict__ out, int n4) {
    int idx = blockIdx.x * blockDim.x + threadIdx.x;
    if (idx >= n4) return;
    v4i v = ((const v4i*)in)[idx];
    unsigned int p = ((unsigned)v.x & 0xFFu) | (((unsigned)v.y & 0xFFu) << 8) |
                     (((unsigned)v.z & 0xFFu) << 16) | (((unsigned)v.w & 0xFFu) << 24);
    out[idx] = (int)p;
}

__global__ __launch_bounds__(256) void zero_i32_kernel(int* __restrict__ p, int n) {
    int idx = blockIdx.x * blockDim.x + threadIdx.x;
    if (idx < n) p[idx] = 0;
}

template <int MODE>
__global__ __launch_bounds__(NTHREADS) void gemm_i8_kernel(
    const u8* __restrict__ Aq,
    const u8* __restrict__ Bq,
    const float* __restrict__ rs,
    const float* __restrict__ cs,
    const float* __restrict__ bias,
    _Float16* __restrict__ g16,
    float* __restrict__ out_f32,
    int* __restrict__ amax_g,
    int M, int N, int K) {
    __shared__ v4i sA4[2 * BLK_M * (BLK_K / 16)];
    __shared__ v4i sB4[2 * BLK_N * (BLK_K / 16)];
    __shared__ int s_amax[BLK_M];

    const int tid  = threadIdx.x;
    const int lane = tid & 31;
    const int wid  = tid >> 5;
    const int wm   = wid >> 1;
    const int wn   = wid & 1;
    const int h    = lane >> 4;
    const int l16  = lane & 15;

    const int n0 = blockIdx.x * BLK_N;
    const int m0 = blockIdx.y * BLK_M;

    if (MODE == 0 && tid < BLK_M) s_amax[tid] = 0;

    v8i acc[4];
#pragma unroll
    for (int st = 0; st < 4; ++st) acc[st] = (v8i){};

    const u64* sA8  = (const u64*)sA4;
    const u32 sA_lo = (u32)(uintptr_t)(&sA4[0]);
    const u32 sB_lo = (u32)(uintptr_t)(&sB4[0]);

    const int a_row = tid >> 2, a_seg = tid & 3;
    const int b_row = tid >> 1, b_half = tid & 1;

    auto issue_stage = [&](int buf, int k0) {
        {
            const u8* g = Aq + (size_t)(m0 + a_row) * K + k0 + a_seg * 32;
            u32 l = sA_lo + (u32)(buf * (BLK_M * BLK_K) + a_row * BLK_K + a_seg * 32);
            async_load_b128(l, g);
            async_load_b128(l + 16, g + 16);
        }
        {
            const u8* g = Bq + (size_t)(n0 + b_row) * K + k0 + b_half * 64;
            u32 l = sB_lo + (u32)(buf * (BLK_N * BLK_K) + b_row * BLK_K + b_half * 64);
            async_load_b128(l, g);
            async_load_b128(l + 16, g + 16);
            async_load_b128(l + 32, g + 32);
            async_load_b128(l + 48, g + 48);
        }
    };

    auto compute = [&](int buf) {
        const int aoff = buf * 1024;
        const int boff = buf * 1024;
        union UA { u64 q[4]; v8i v; } ua[2];
        const int abase = aoff + (wm * 16 + l16) * 16;
#pragma unroll
        for (int kk = 0; kk < 2; ++kk) {
            ua[kk].q[0] = sA8[abase + kk * 8 + 0 + 2 * h];
            ua[kk].q[1] = sA8[abase + kk * 8 + 1 + 2 * h];
            ua[kk].q[2] = sA8[abase + kk * 8 + 4 + 2 * h];
            ua[kk].q[3] = sA8[abase + kk * 8 + 5 + 2 * h];
        }
#pragma unroll
        for (int st = 0; st < 4; ++st) {
            const int bbase = boff + (wn * 64 + st * 16 + l16) * 8;
            union UB { v4i p[2]; v8i v; } ub0, ub1;
            ub0.p[0] = sB4[bbase + 0 + h];
            ub0.p[1] = sB4[bbase + 2 + h];
            ub1.p[0] = sB4[bbase + 4 + h];
            ub1.p[1] = sB4[bbase + 6 + h];
            acc[st] = __builtin_amdgcn_wmma_i32_16x16x64_iu8(
                true, ua[0].v, true, ub0.v, acc[st], false, false);
            acc[st] = __builtin_amdgcn_wmma_i32_16x16x64_iu8(
                true, ua[1].v, true, ub1.v, acc[st], false, false);
        }
    };

    const int KT = K / BLK_K;
    issue_stage(0, 0);

    int kt = 0;
    for (;;) {
        if (kt + 1 < KT) { issue_stage(1, (kt + 1) * BLK_K); wait_async_le6(); }
        else             { wait_async_0(); }
        __syncthreads();
        compute(0);
        __syncthreads();
        if (++kt == KT) break;

        if (kt + 1 < KT) { issue_stage(0, (kt + 1) * BLK_K); wait_async_le6(); }
        else             { wait_async_0(); }
        __syncthreads();
        compute(1);
        __syncthreads();
        if (++kt == KT) break;
    }

#pragma unroll
    for (int r = 0; r < 8; ++r) {
        const int m_l = wm * 16 + r + 8 * h;
        const int t   = m0 + m_l;
        if (MODE == 0) {
            const float rscale = rs[t];
            float rowmax = 0.0f;
#pragma unroll
            for (int st = 0; st < 4; ++st) {
                int n   = n0 + wn * 64 + st * 16 + l16;
                float v = (float)acc[st][r] * rscale * cs[n] + bias[n];
                float f = (float)(_Float16)v;
                float g = 0.5f * f * (1.0f + erff(f * 0.70710678118654752f));
                g16[(size_t)t * N + n] = (_Float16)g;
                rowmax = fmaxf(rowmax, fabsf(g));
            }
            atomicMax(&s_amax[m_l], __float_as_int(rowmax));
        } else {
            const float rscale = rs[t] * (1.0f / 127.0f);
#pragma unroll
            for (int st = 0; st < 4; ++st) {
                int n   = n0 + wn * 64 + st * 16 + l16;
                float v = (float)acc[st][r] * rscale * cs[n] + bias[n];
                out_f32[(size_t)t * N + n] = (float)(_Float16)v;
            }
        }
    }
    if (MODE == 0) {
        __syncthreads();
        if (tid < BLK_M) atomicMax(amax_g + m0 + tid, s_amax[tid]);
    }
}

__global__ __launch_bounds__(256) void quant_kernel(const _Float16* __restrict__ g16,
                                                    const float* __restrict__ amax,
                                                    int* __restrict__ act8, int n4) {
    int idx = blockIdx.x * blockDim.x + threadIdx.x;
    if (idx >= n4) return;
    int t = idx / (I_DIM / 4);
    union { u64 u; _Float16 hv[4]; } g;
    g.u = ((const u64*)g16)[idx];
    float am  = amax[t];
    float inv = (am > 0.0f) ? (127.0f / am) : 0.0f;
    unsigned int p = 0;
#pragma unroll
    for (int j = 0; j < 4; ++j) {
        float q = rintf((float)g.hv[j] * inv);
        q = fminf(127.0f, fmaxf(-127.0f, q));
        p |= ((unsigned)(int)q & 0xFFu) << (8 * j);
    }
    act8[idx] = (int)p;
}

extern "C" void kernel_launch(void* const* d_in, const int* in_sizes, int n_in,
                              void* d_out, int out_size, void* d_ws, size_t ws_size,
                              hipStream_t stream) {
    const int*   hs  = (const int*)d_in[0];
    const float* sc  = (const float*)d_in[1];
    const int*   w1  = (const int*)d_in[2];
    const float* w1s = (const float*)d_in[3];
    const float* b1  = (const float*)d_in[4];
    const int*   w2  = (const int*)d_in[5];
    const float* w2s = (const float*)d_in[6];
    const float* b2  = (const float*)d_in[7];
    float* out = (float*)d_out;

    char* ws = (char*)d_ws;
    u8*  x8    = (u8*)(ws);
    u8*  w1_8  = (u8*)(ws + 26214400);
    u8*  w2_8  = (u8*)(ws + 67174400);
    u8*  act8  = (u8*)(ws + 108134400);
    _Float16* g16 = (_Float16*)(ws + 212992000);
    int* amax  = (int*)(ws + 422707200);

    const int n4x = T_TOK * H_DIM / 4;
    const int n4w = I_DIM * H_DIM / 4;
    const int n4a = T_TOK * I_DIM / 4;

    pack_i8_kernel<<<(n4x + 255) / 256, 256, 0, stream>>>(hs, (int*)x8, n4x);
    pack_i8_kernel<<<(n4w + 255) / 256, 256, 0, stream>>>(w1, (int*)w1_8, n4w);
    pack_i8_kernel<<<(n4w + 255) / 256, 256, 0, stream>>>(w2, (int*)w2_8, n4w);
    zero_i32_kernel<<<(T_TOK + 255) / 256, 256, 0, stream>>>(amax, T_TOK);

    gemm_i8_kernel<0><<<dim3(I_DIM / BLK_N, T_TOK / BLK_M), NTHREADS, 0, stream>>>(
        x8, w1_8, sc, w1s, b1, g16, nullptr, amax, T_TOK, I_DIM, H_DIM);

    quant_kernel<<<(n4a + 255) / 256, 256, 0, stream>>>(g16, (const float*)amax,
                                                        (int*)act8, n4a);

    gemm_i8_kernel<1><<<dim3(H_DIM / BLK_N, T_TOK / BLK_M), NTHREADS, 0, stream>>>(
        act8, w2_8, (const float*)amax, w2s, b2, nullptr, out, nullptr,
        T_TOK, H_DIM, I_DIM);
}
